// MultiHeadAttention_19954418057255
// MI455X (gfx1250) — matrix operand order repaired and checked by reading, not compiled, not run on hardware
//
#include <hip/hip_runtime.h>

// Problem constants (match reference: B=2, T=2048, E=1024, H=16, DK=64)
#define BB   2
#define TT   2048
#define EE   1024
#define HH   16
#define DK   64
#define MR   (BB*TT)   // 4096 rows in the flattened [B*T, E] activations

#ifndef __has_builtin
#define __has_builtin(x) 0
#endif

#if __has_builtin(__builtin_amdgcn_global_load_async_to_lds_b128)
#define HAVE_ASYNC_LDS 1
#else
#define HAVE_ASYNC_LDS 0
#endif

typedef __attribute__((ext_vector_type(16))) __bf16 v16bf;
typedef __attribute__((ext_vector_type(8)))  __bf16 v8bf;
typedef __attribute__((ext_vector_type(8)))  float  v8f;

// Pointer types matching the async-LDS builtin signature:
// (int4 addrspace(1)*, int4 addrspace(3)*, imm int, imm int)
typedef int b128i __attribute__((vector_size(16)));
typedef __attribute__((address_space(1))) b128i* as1_b128;
typedef __attribute__((address_space(3))) b128i* as3_b128;

// ---------------------------------------------------------------------------
// WMMA fragment helpers (layouts per CDNA5 ISA 7.12.2, wave32)
// ---------------------------------------------------------------------------

// A fragment: 16x32 bf16 (MxK). Lane l: row m = l&15, half = l>>4.
// e in [0,8):  K = half*8 + e          e in [8,16): K = 16 + half*8 + (e-8)
static __device__ __forceinline__ v16bf load_a_frag(const __bf16* A, int lda) {
  const int lane = threadIdx.x & 31;
  const int m = lane & 15, h = lane >> 4;
  const __bf16* row = A + (size_t)m * lda + h * 8;
  v8bf lo = *(const v8bf*)(row);
  v8bf hi = *(const v8bf*)(row + 16);
  v16bf a;
#pragma unroll
  for (int e = 0; e < 8; ++e) { a[e] = lo[e]; a[e + 8] = hi[e]; }
  return a;
}

// B fragment where mathematical B = W^T with W row-major [N][K]:
// b[e] = W[n][half*8 + e] for e<8, W[n][16 + half*8 + (e-8)] for e>=8
static __device__ __forceinline__ v16bf load_bT_frag(const __bf16* W, int ldw) {
  const int lane = threadIdx.x & 31;
  const int n = lane & 15, h = lane >> 4;
  const __bf16* p = W + (size_t)n * ldw + h * 8;
  v8bf lo = *(const v8bf*)(p);
  v8bf hi = *(const v8bf*)(p + 16);
  v16bf b;
#pragma unroll
  for (int e = 0; e < 8; ++e) { b[e] = lo[e]; b[e + 8] = hi[e]; }
  return b;
}

static __device__ __forceinline__ v8f wmma_bf16(v16bf a, v16bf b, v8f c) {
  return __builtin_amdgcn_wmma_f32_16x16x32_bf16(false, a, false, b, (short)0, c,
                                                 false, false);
}

// Async global -> LDS 16-byte copy (CDNA5 ASYNCcnt path), with sync fallback.
static __device__ __forceinline__ void async_copy_b128(const __bf16* g, __bf16* l) {
#if HAVE_ASYNC_LDS
  __builtin_amdgcn_global_load_async_to_lds_b128((as1_b128)g, (as3_b128)l, 0, 0);
#else
  *(v8bf*)l = *(const v8bf*)g;
#endif
}

static __device__ __forceinline__ void wait_async() {
#if HAVE_ASYNC_LDS
#if __has_builtin(__builtin_amdgcn_s_wait_asynccnt)
  __builtin_amdgcn_s_wait_asynccnt(0);
#else
  asm volatile("s_wait_asynccnt 0" ::: "memory");
#endif
#endif
}

// ---------------------------------------------------------------------------
// f32 -> bf16 conversion
// ---------------------------------------------------------------------------
__global__ void cvt_kernel(const float* __restrict__ in, __bf16* __restrict__ out,
                           int n) {
  int i = blockIdx.x * blockDim.x + threadIdx.x;
  int stride = gridDim.x * blockDim.x;
  for (; i < n; i += stride) out[i] = (__bf16)in[i];
}

// ---------------------------------------------------------------------------
// QKV projection GEMM:  Y = (X @ W^T + b) * scale  -> bf16
//   vtrans == 0 : Y stored [B,H,T,DK]   (for Q, K)
//   vtrans == 1 : Y stored [B,H,DK,T]   (for V: makes P@V B-fragments contiguous)
// X: [MR, EE] bf16 row-major; W: [EE, EE] bf16 row-major (N=row, K=col)
// Workgroup 256 thr = 8 waves (4 along M x 2 along N); wave tile 32x64.
// ---------------------------------------------------------------------------
__global__ void __launch_bounds__(256) qkv_gemm_kernel(
    const __bf16* __restrict__ X, const __bf16* __restrict__ W,
    const float* __restrict__ bias, __bf16* __restrict__ Out, float scale,
    int vtrans) {
  const int w = threadIdx.x >> 5;
  const int m0 = blockIdx.x * 128 + (w >> 1) * 32;
  const int n0 = blockIdx.y * 128 + (w & 1) * 64;

  v8f acc[2][4];
#pragma unroll
  for (int i = 0; i < 2; ++i)
#pragma unroll
    for (int j = 0; j < 4; ++j) acc[i][j] = (v8f){};

  for (int kk = 0; kk < EE; kk += 32) {
    v16bf a0 = load_a_frag(X + (size_t)m0 * EE + kk, EE);
    v16bf a1 = load_a_frag(X + (size_t)(m0 + 16) * EE + kk, EE);
#pragma unroll
    for (int j = 0; j < 4; ++j) {
      v16bf b = load_bT_frag(W + (size_t)(n0 + j * 16) * EE + kk, EE);
      acc[0][j] = wmma_bf16(a0, b, acc[0][j]);
      acc[1][j] = wmma_bf16(a1, b, acc[1][j]);
    }
  }

  const int lane = threadIdx.x & 31;
  const int nl = lane & 15, hf = lane >> 4;
#pragma unroll
  for (int j = 0; j < 4; ++j) {
    const int n = n0 + j * 16 + nl;
    const float bn = bias[n];
    const int hh = n >> 6, d = n & 63;
#pragma unroll
    for (int i = 0; i < 2; ++i) {
#pragma unroll
      for (int r = 0; r < 8; ++r) {
        const int m = m0 + i * 16 + r + 8 * hf;
        const int bb = m >> 11;           // /TT
        const int t  = m & (TT - 1);
        float v = (acc[i][j][r] + bn) * scale;
        if (vtrans)
          Out[(((size_t)bb * HH + hh) * DK + d) * TT + t] = (__bf16)v;
        else
          Out[(((size_t)bb * HH + hh) * TT + t) * DK + d] = (__bf16)v;
      }
    }
  }
}

// ---------------------------------------------------------------------------
// Flash attention with the reference's strictly-upper-triangular keep mask.
// Q,K: bf16 [B,H,T,DK] (Q pre-scaled by 1/sqrt(DK)); Vt: bf16 [B,H,DK,T].
// O: bf16 [B,T,E].
// grid = (T/128, H, B); block = 256 (8 waves, one 16-row q-tile per wave).
// All 8 waves share the same (b,h) and key schedule, so K/V tiles are staged
// cooperatively into LDS with double-buffered async copies.
// ---------------------------------------------------------------------------
__global__ void __launch_bounds__(256) attn_kernel(const __bf16* __restrict__ Q,
                                                   const __bf16* __restrict__ K,
                                                   const __bf16* __restrict__ Vt,
                                                   __bf16* __restrict__ O) {
  __shared__ __align__(16) __bf16 kbuf[2][32][72];  // K tile: 32 keys x 64 d (padded)
  __shared__ __align__(16) __bf16 vbuf[2][64][40];  // Vt tile: 64 d x 32 keys (padded)
  __shared__ __align__(16) __bf16 pbuf[8][16][40];  // per-wave P re-layout tile

  const int tid  = threadIdx.x;
  const int w    = tid >> 5;
  const int lane = tid & 31;
  const int nl = lane & 15, hf = lane >> 4;
  const int q0 = blockIdx.x * 128 + w * 16;
  const size_t hb = ((size_t)blockIdx.z * HH + blockIdx.y) * TT * DK;
  const __bf16* Qh = Q + hb;
  const __bf16* Kh = K + hb;
  const __bf16* Vh = Vt + hb;   // row-major [DK][TT] within the head

  // Cooperative stage of one 32-key tile (K: 32x64, Vt: 64x32) into LDS.
  // 256 threads x one b128 each per matrix.
  auto stage = [&](int buf, int tk) {
    {
      const int r = tid >> 3, c = tid & 7;                 // 32 rows x 8 chunks
      async_copy_b128(Kh + (size_t)(tk + r) * DK + c * 8, &kbuf[buf][r][c * 8]);
    }
    {
      const int r = tid >> 2, c = tid & 3;                 // 64 rows x 4 chunks
      async_copy_b128(Vh + (size_t)r * TT + tk + c * 8, &vbuf[buf][r][c * 8]);
    }
  };

  // Resident Q fragments (16 rows x 64 K, split into two K=32 fragments)
  v16bf qa0 = load_a_frag(Qh + (size_t)q0 * DK, DK);
  v16bf qa1 = load_a_frag(Qh + (size_t)q0 * DK + 32, DK);

  v8f o[4];
#pragma unroll
  for (int j = 0; j < 4; ++j) o[j] = (v8f){};
  float mrow[8], lrow[8];
#pragma unroll
  for (int r = 0; r < 8; ++r) { mrow[r] = -1e30f; lrow[r] = 0.0f; }

  stage(0, 0);

  const int NT = TT / 32;
  for (int it = 0; it < NT; ++it) {
    const int tk = it * 32;
    const int buf = it & 1;

    wait_async();        // this wave's copies of tile `it` (incl. prefetch) done
    __syncthreads();     // all waves' copies visible; prior tile fully consumed

    if (it + 1 < NT) stage(buf ^ 1, tk + 32);  // overlap next copy with compute

    const __bf16* kt = &kbuf[buf][0][0];
    const __bf16* vt = &vbuf[buf][0][0];

    // S tile 16x32 = two 16x16 accumulators (ds_load_b128 fragment reads)
    v16bf kb;
    v8f s0 = (v8f){}, s1 = (v8f){};
    kb = load_bT_frag(kt, 72);            s0 = wmma_bf16(qa0, kb, s0);
    kb = load_bT_frag(kt + 32, 72);       s0 = wmma_bf16(qa1, kb, s0);
    kb = load_bT_frag(kt + 16 * 72, 72);      s1 = wmma_bf16(qa0, kb, s1);
    kb = load_bT_frag(kt + 16 * 72 + 32, 72); s1 = wmma_bf16(qa1, kb, s1);

#pragma unroll
    for (int r = 0; r < 8; ++r) {
      const int qg = q0 + r + 8 * hf;
      // Reference bug preserved: keep only strictly-future keys (k > q)
      float v0 = ((tk + nl)      > qg) ? s0[r] : -1e9f;
      float v1 = ((tk + 16 + nl) > qg) ? s1[r] : -1e9f;

      float tm = fmaxf(v0, v1);
      tm = fmaxf(tm, __shfl_xor(tm, 1, 32));
      tm = fmaxf(tm, __shfl_xor(tm, 2, 32));
      tm = fmaxf(tm, __shfl_xor(tm, 4, 32));
      tm = fmaxf(tm, __shfl_xor(tm, 8, 32));

      const float mnew  = fmaxf(mrow[r], tm);
      const float alpha = __expf(mrow[r] - mnew);
      const float p0 = __expf(v0 - mnew);
      const float p1 = __expf(v1 - mnew);
      float rs = p0 + p1;
      rs += __shfl_xor(rs, 1, 32);
      rs += __shfl_xor(rs, 2, 32);
      rs += __shfl_xor(rs, 4, 32);
      rs += __shfl_xor(rs, 8, 32);
      lrow[r] = lrow[r] * alpha + rs;
      mrow[r] = mnew;
#pragma unroll
      for (int j = 0; j < 4; ++j) o[j][r] *= alpha;

      // Stage P (C layout) into LDS for re-read in A layout
      pbuf[w][r + 8 * hf][nl]      = (__bf16)p0;
      pbuf[w][r + 8 * hf][16 + nl] = (__bf16)p1;
    }

    // Wave-private LDS tile: waves are lockstep; just drain DS ops.
    asm volatile("s_wait_dscnt 0" ::: "memory");

    // Re-read P as an A fragment (16x32)
    v16bf pa;
    {
      const __bf16* prow = &pbuf[w][nl][hf * 8];
      v8bf lo = *(const v8bf*)(prow);
      v8bf hi = *(const v8bf*)(prow + 16);
#pragma unroll
      for (int e = 0; e < 8; ++e) { pa[e] = lo[e]; pa[e + 8] = hi[e]; }
    }

    // O += P @ V : B fragment b[e] = Vt[d = j*16 + n][tk + half*8 + e + (e >= 8 ? 8 : 0)]
#pragma unroll
    for (int j = 0; j < 4; ++j) {
      v16bf vb = load_bT_frag(vt + (size_t)(j * 16) * 40, 40);
      o[j] = wmma_bf16(pa, vb, o[j]);
    }
  }

  // Normalize and store O as bf16 in [B, T, E] (e = head*DK + d)
#pragma unroll
  for (int r = 0; r < 8; ++r) {
    const float inv = 1.0f / lrow[r];
    const int t = q0 + r + 8 * hf;
    const size_t base =
        ((size_t)blockIdx.z * TT + t) * EE + (size_t)blockIdx.y * DK;
#pragma unroll
    for (int j = 0; j < 4; ++j)
      O[base + j * 16 + nl] = (__bf16)(o[j][r] * inv);
  }
}

// ---------------------------------------------------------------------------
// Output projection:  out = Obf @ Wo^T + bo   (f32 output, row-major [MR, EE])
// ---------------------------------------------------------------------------
__global__ void __launch_bounds__(256) out_gemm_kernel(
    const __bf16* __restrict__ X, const __bf16* __restrict__ W,
    const float* __restrict__ bias, float* __restrict__ Y) {
  const int w = threadIdx.x >> 5;
  const int m0 = blockIdx.x * 128 + (w >> 1) * 32;
  const int n0 = blockIdx.y * 128 + (w & 1) * 64;

  v8f acc[2][4];
#pragma unroll
  for (int i = 0; i < 2; ++i)
#pragma unroll
    for (int j = 0; j < 4; ++j) acc[i][j] = (v8f){};

  for (int kk = 0; kk < EE; kk += 32) {
    v16bf a0 = load_a_frag(X + (size_t)m0 * EE + kk, EE);
    v16bf a1 = load_a_frag(X + (size_t)(m0 + 16) * EE + kk, EE);
#pragma unroll
    for (int j = 0; j < 4; ++j) {
      v16bf b = load_bT_frag(W + (size_t)(n0 + j * 16) * EE + kk, EE);
      acc[0][j] = wmma_bf16(a0, b, acc[0][j]);
      acc[1][j] = wmma_bf16(a1, b, acc[1][j]);
    }
  }

  const int lane = threadIdx.x & 31;
  const int nl = lane & 15, hf = lane >> 4;
#pragma unroll
  for (int j = 0; j < 4; ++j) {
    const int n = n0 + j * 16 + nl;
    const float bn = bias[n];
#pragma unroll
    for (int i = 0; i < 2; ++i) {
#pragma unroll
      for (int r = 0; r < 8; ++r) {
        const int m = m0 + i * 16 + r + 8 * hf;
        Y[(size_t)m * EE + n] = acc[i][j][r] + bn;
      }
    }
  }
}

// ---------------------------------------------------------------------------
// Host launcher
// ---------------------------------------------------------------------------
extern "C" void kernel_launch(void* const* d_in, const int* in_sizes, int n_in,
                              void* d_out, int out_size, void* d_ws,
                              size_t ws_size, hipStream_t stream) {
  (void)in_sizes; (void)n_in; (void)out_size; (void)ws_size;
  const float* x  = (const float*)d_in[0];
  const float* Wq = (const float*)d_in[1];
  const float* bq = (const float*)d_in[2];
  const float* Wk = (const float*)d_in[3];
  const float* bk = (const float*)d_in[4];
  const float* Wv = (const float*)d_in[5];
  const float* bv = (const float*)d_in[6];
  const float* Wo = (const float*)d_in[7];
  const float* bo = (const float*)d_in[8];
  float* out = (float*)d_out;

  const size_t NX = (size_t)MR * EE;  // 4,194,304
  const size_t NW = (size_t)EE * EE;  // 1,048,576

  __bf16* xbf = (__bf16*)d_ws;
  __bf16* wqb = xbf + NX;
  __bf16* wkb = wqb + NW;
  __bf16* wvb = wkb + NW;
  __bf16* wob = wvb + NW;
  __bf16* qbf = wob + NW;   // [B,H,T,DK], pre-scaled by 0.125
  __bf16* kbf = qbf + NX;   // [B,H,T,DK]
  __bf16* vbf = kbf + NX;   // [B,H,DK,T]  (transposed)
  __bf16* obf = vbf + NX;   // [B,T,E]

  cvt_kernel<<<2048, 256, 0, stream>>>(x,  xbf, (int)NX);
  cvt_kernel<<<1024, 256, 0, stream>>>(Wq, wqb, (int)NW);
  cvt_kernel<<<1024, 256, 0, stream>>>(Wk, wkb, (int)NW);
  cvt_kernel<<<1024, 256, 0, stream>>>(Wv, wvb, (int)NW);
  cvt_kernel<<<1024, 256, 0, stream>>>(Wo, wob, (int)NW);

  dim3 gGemm(MR / 128, EE / 128);  // 32 x 8
  qkv_gemm_kernel<<<gGemm, 256, 0, stream>>>(xbf, wqb, bq, qbf, 0.125f, 0);
  qkv_gemm_kernel<<<gGemm, 256, 0, stream>>>(xbf, wkb, bk, kbf, 1.0f, 0);
  qkv_gemm_kernel<<<gGemm, 256, 0, stream>>>(xbf, wvb, bv, vbf, 1.0f, 1);

  dim3 gAttn(TT / 128, HH, BB);  // 16 x 16 x 2
  attn_kernel<<<gAttn, 256, 0, stream>>>(qbf, kbf, vbf, obf);

  out_gemm_kernel<<<gGemm, 256, 0, stream>>>(obf, wob, bo, out);
}
